// SwitchableQATGPT2_65429531787891
// MI455X (gfx1250) — compile-verified, matrix operand order checked by reading, not run on hardware
//
#include <hip/hip_runtime.h>

// ---------------------------------------------------------------------------
// Types
// ---------------------------------------------------------------------------
typedef __attribute__((ext_vector_type(16))) __bf16 v16bf;
typedef __attribute__((ext_vector_type(8)))  float  v8f;
typedef unsigned int u32;
typedef __attribute__((ext_vector_type(4))) u32 u32x4;
typedef __attribute__((ext_vector_type(8))) int i32x8;
typedef __attribute__((ext_vector_type(4))) int i32x4;

struct Frag256 { uint4 lo; uint4 hi; };   // 32 bytes == v16bf

__device__ __forceinline__ unsigned short qat_f2bf(float f) {
    unsigned u = __builtin_bit_cast(unsigned, f);
    unsigned r = (u + 0x7FFFu + ((u >> 16) & 1u)) >> 16;   // RNE
    return (unsigned short)r;
}

// ---------------------------------------------------------------------------
// TDM: 2D tile (tile_d0 contiguous elems x tile_d1 rows) of fp32 from global
// to LDS. Zero-fills out-of-range reads (tensor_d0/d1 are remaining extents).
// D# layout per CDNA5 ISA:
//   g0: [1:0]=count=1, [63:32]=lds_addr(bytes), [120:64]=global_addr,
//       [127:126]=type=2
//   g1: [17:16]=data_size(2 -> 4B), [79:48]=tensor_dim0, [111:80]=tensor_dim1,
//       [127:112]=tile_dim0, [143:128]=tile_dim1, [207:160]=tensor_dim0_stride
// This toolchain exposes the 6-arg builtin: (g0, g1, g2, g3, g4, cpol);
// groups 2..4 zero-filled (2D tensor -> higher-dim groups disabled).
// ---------------------------------------------------------------------------
__device__ __forceinline__ void qat_tdm_load_2d(
    u32 lds_byte_off, const void* gptr, u32 tensor_d0, u32 tensor_d1,
    u32 tile_d0, u32 tile_d1, unsigned long long stride_elems)
{
    unsigned long long ga = (unsigned long long)gptr;
    u32x4 g0;
    g0.x = 1u;                                           // count=1
    g0.y = lds_byte_off;                                 // lds_addr
    g0.z = (u32)ga;                                      // global_addr[31:0]
    g0.w = ((u32)(ga >> 32) & 0x01FFFFFFu) | (2u << 30); // addr[56:32] | type=2
    i32x8 g1;
    g1[0] = (int)(2u << 16);                             // data_size = 4 bytes
    g1[1] = (int)((tensor_d0 & 0xFFFFu) << 16);          // tensor_dim0[15:0]
    g1[2] = (int)((tensor_d0 >> 16) | ((tensor_d1 & 0xFFFFu) << 16));
    g1[3] = (int)((tensor_d1 >> 16) | (tile_d0 << 16));  // tile_dim0
    g1[4] = (int)(tile_d1 & 0xFFFFu);                    // tile_dim1 (tile_dim2=0)
    g1[5] = (int)(u32)(stride_elems & 0xFFFFFFFFull);    // dim0_stride[31:0]
    g1[6] = (int)(u32)((stride_elems >> 32) & 0xFFFFull);// dim0_stride[47:32]
    g1[7] = 0;
    i32x4 z4 = {0, 0, 0, 0};
    i32x8 z8 = {0, 0, 0, 0, 0, 0, 0, 0};
    __builtin_amdgcn_tensor_load_to_lds(g0, g1, z4, z4, z8, 0);
}

__device__ __forceinline__ v16bf qat_load_frag(const unsigned short* tileBase, int lane) {
    int m = lane & 15, g = lane >> 4;
    const uint4* p = (const uint4*)(tileBase + m * 32);
    Frag256 u;
    u.lo = p[g];
    u.hi = p[2 + g];
    return __builtin_bit_cast(v16bf, u);
}

// ---------------------------------------------------------------------------
// GEMM: C[M,N] = scale * ( bf16(A) @ bf16(fq(W)) ) + bias
// Assumes M % 128 == 0 and K % 32 == 0 (true for every launch here);
// the N edge is zero-filled by TDM on loads and guarded on store.
// ---------------------------------------------------------------------------
__global__ __launch_bounds__(256) void qat_gemm_wmma(
    const float* __restrict__ A, const float* __restrict__ W,
    const float* __restrict__ maxabs_slot, const float* __restrict__ bias,
    float* __restrict__ C, int M, int N, int K, int ldw, int wTrans, int quant)
{
    __shared__ float          As32[128 * 32];   // fp32 stage: [row][k]
    __shared__ float          Bs32[128 * 32];   // fp32 stage: [k][n] or [n][k]
    __shared__ unsigned short As16[128 * 32];   // bf16 [row][k]
    __shared__ unsigned short Bs16[128 * 32];   // bf16 [n][k]

    const int tid  = threadIdx.x;
    const int lane = tid & 31;
    const int wave = tid >> 5;
    const int wm   = wave >> 1;     // 0..3
    const int wn   = wave & 1;      // 0..1
    const int bM   = blockIdx.y * 128;
    const int bN   = blockIdx.x * 128;

    float s = 1.0f, inv_s = 1.0f;
    if (quant) {
        float ma = *maxabs_slot;
        s = fmaxf(ma * (1.0f / 127.0f), 1e-8f);
        inv_s = 1.0f / s;
    }
    const bool hasBias = (bias != nullptr);

    const u32 as_off = (u32)(unsigned long long)(const void*)As32;
    const u32 bs_off = (u32)(unsigned long long)(const void*)Bs32;

    v8f acc[2][4] = {};

    for (int k0 = 0; k0 < K; k0 += 32) {
        // -------- async tile DMA into LDS (one wave issues; TDM ignores EXEC)
        if (wave == 0) {
            // A tile: 32 contiguous k x 128 rows, stride K
            qat_tdm_load_2d(as_off, A + (size_t)bM * K + k0,
                            (u32)(K - k0), (u32)(M - bM), 32u, 128u,
                            (unsigned long long)K);
            if (!wTrans) {
                // W tile: 128 contiguous n x 32 k-rows, stride ldw; OOB n -> 0
                qat_tdm_load_2d(bs_off, W + (size_t)k0 * ldw + bN,
                                (u32)(N - bN), (u32)(K - k0), 128u, 32u,
                                (unsigned long long)ldw);
            } else {
                // W^T tile: 32 contiguous k x 128 n-rows, stride ldw; OOB n -> 0
                qat_tdm_load_2d(bs_off, W + (size_t)bN * ldw + k0,
                                (u32)(K - k0), (u32)(N - bN), 32u, 128u,
                                (unsigned long long)ldw);
            }
            __builtin_amdgcn_s_wait_tensorcnt(0);
        }
        if (k0 + 32 < K)
            __builtin_prefetch(&A[(size_t)(bM + (tid & 127)) * K + k0 + 32], 0, 1);
        __syncthreads();

        // -------- convert fp32 LDS -> bf16 LDS (fused weight fake-quant)
        #pragma unroll
        for (int i = 0; i < 4; ++i) {
            int f = tid + i * 256;             // float4 unit 0..1023
            int row = f >> 3, c4 = (f & 7) * 4;
            float4 v = *(const float4*)&As32[row * 32 + c4];
            ushort4 o;
            o.x = qat_f2bf(v.x); o.y = qat_f2bf(v.y);
            o.z = qat_f2bf(v.z); o.w = qat_f2bf(v.w);
            *(ushort4*)&As16[row * 32 + c4] = o;
        }
        if (wTrans) {
            #pragma unroll
            for (int i = 0; i < 4; ++i) {      // Bs32 is [n][k]: same fast path
                int f = tid + i * 256;
                int row = f >> 3, c4 = (f & 7) * 4;
                float4 v = *(const float4*)&Bs32[row * 32 + c4];
                if (quant) {
                    v.x = fminf(fmaxf(rintf(v.x * inv_s), -128.0f), 127.0f);
                    v.y = fminf(fmaxf(rintf(v.y * inv_s), -128.0f), 127.0f);
                    v.z = fminf(fmaxf(rintf(v.z * inv_s), -128.0f), 127.0f);
                    v.w = fminf(fmaxf(rintf(v.w * inv_s), -128.0f), 127.0f);
                }
                ushort4 o;
                o.x = qat_f2bf(v.x); o.y = qat_f2bf(v.y);
                o.z = qat_f2bf(v.z); o.w = qat_f2bf(v.w);
                *(ushort4*)&Bs16[row * 32 + c4] = o;
            }
        } else {
            #pragma unroll
            for (int i = 0; i < 4; ++i) {      // Bs32 is [k][n]: transpose stage
                int f = tid + i * 256;
                int k = f >> 5, n4 = (f & 31) * 4;
                float4 v = *(const float4*)&Bs32[k * 128 + n4];
                if (quant) {
                    v.x = fminf(fmaxf(rintf(v.x * inv_s), -128.0f), 127.0f);
                    v.y = fminf(fmaxf(rintf(v.y * inv_s), -128.0f), 127.0f);
                    v.z = fminf(fmaxf(rintf(v.z * inv_s), -128.0f), 127.0f);
                    v.w = fminf(fmaxf(rintf(v.w * inv_s), -128.0f), 127.0f);
                }
                Bs16[(n4 + 0) * 32 + k] = qat_f2bf(v.x);
                Bs16[(n4 + 1) * 32 + k] = qat_f2bf(v.y);
                Bs16[(n4 + 2) * 32 + k] = qat_f2bf(v.z);
                Bs16[(n4 + 3) * 32 + k] = qat_f2bf(v.w);
            }
        }
        __syncthreads();

        // -------- fragments + WMMA
        v16bf af[2], bf[4];
        #pragma unroll
        for (int tm = 0; tm < 2; ++tm)
            af[tm] = qat_load_frag(As16 + (wm * 32 + tm * 16) * 32, lane);
        #pragma unroll
        for (int tn = 0; tn < 4; ++tn)
            bf[tn] = qat_load_frag(Bs16 + (wn * 64 + tn * 16) * 32, lane);

        #pragma unroll
        for (int tm = 0; tm < 2; ++tm)
            #pragma unroll
            for (int tn = 0; tn < 4; ++tn)
                acc[tm][tn] = __builtin_amdgcn_wmma_f32_16x16x32_bf16(
                    false, af[tm], false, bf[tn], (short)0, acc[tm][tn], false, false);
        __syncthreads();
    }

    // C/D layout: VGPR r, lane L: M = r + 8*(L/16), N = L%16
    const int g  = lane >> 4;
    const int nl = lane & 15;
    #pragma unroll
    for (int tm = 0; tm < 2; ++tm)
        #pragma unroll
        for (int tn = 0; tn < 4; ++tn)
            #pragma unroll
            for (int r = 0; r < 8; ++r) {
                int row = bM + wm * 32 + tm * 16 + r + 8 * g;
                int col = bN + wn * 64 + tn * 16 + nl;
                if (col < N) {
                    float v = acc[tm][tn][r] * s;
                    if (hasBias) v += bias[col];
                    C[(size_t)row * N + col] = v;
                }
            }
}

// ---------------------------------------------------------------------------
// Small fp32 kernels
// ---------------------------------------------------------------------------
__global__ void qat_init_scales(float* scales) {
    scales[threadIdx.x] = -3.0e38f;
}

__global__ void qat_embed(const int* __restrict__ ids, const float* __restrict__ wte,
                          const float* __restrict__ wpe, float* __restrict__ h) {
    int row = blockIdx.x;                // 0..2047 (b*1024+t)
    int id  = ids[row];
    int pos = row & 1023;
    for (int c = threadIdx.x; c < 768; c += 256)
        h[(size_t)row * 768 + c] = wte[(size_t)id * 768 + c] + wpe[(size_t)pos * 768 + c];
}

__global__ void qat_ln(const float* __restrict__ Hin, const float* __restrict__ gg,
                       const float* __restrict__ bb, float* __restrict__ Xout, int Dn) {
    __shared__ float red[256];
    int row = blockIdx.x, t = threadIdx.x;
    const float* hr = Hin + (size_t)row * Dn;
    float s0 = 0.0f;
    for (int c = t; c < Dn; c += 256) s0 += hr[c];
    red[t] = s0; __syncthreads();
    for (int st = 128; st > 0; st >>= 1) { if (t < st) red[t] += red[t + st]; __syncthreads(); }
    float mean = red[0] / Dn; __syncthreads();
    float s1 = 0.0f;
    for (int c = t; c < Dn; c += 256) { float d = hr[c] - mean; s1 += d * d; }
    red[t] = s1; __syncthreads();
    for (int st = 128; st > 0; st >>= 1) { if (t < st) red[t] += red[t + st]; __syncthreads(); }
    float inv = rsqrtf(red[0] / Dn + 1e-5f);
    for (int c = t; c < Dn; c += 256)
        Xout[(size_t)row * Dn + c] = (hr[c] - mean) * inv * gg[c] + bb[c];
}

__global__ void qat_maxabs(const float* __restrict__ w, size_t n, float* slot) {
    __shared__ float red[256];
    float m = 0.0f;
    for (size_t i = (size_t)blockIdx.x * 256 + threadIdx.x; i < n; i += (size_t)gridDim.x * 256)
        m = fmaxf(m, fabsf(w[i]));
    red[threadIdx.x] = m; __syncthreads();
    for (int st = 128; st > 0; st >>= 1) {
        if (threadIdx.x < st) red[threadIdx.x] = fmaxf(red[threadIdx.x], red[threadIdx.x + st]);
        __syncthreads();
    }
    if (threadIdx.x == 0) atomicMax(slot, red[0]);
}

__global__ void qat_minmax_strided(const float* __restrict__ base, int rows, int cols,
                                   int stride, float* mxslot, float* negmnslot) {
    __shared__ float r1[256], r2[256];
    size_t n = (size_t)rows * cols;
    float mx = -3.0e38f, nm = -3.0e38f;
    for (size_t i = (size_t)blockIdx.x * 256 + threadIdx.x; i < n; i += (size_t)gridDim.x * 256) {
        int r = (int)(i / cols), c = (int)(i % cols);
        float v = base[(size_t)r * stride + c];
        mx = fmaxf(mx, v); nm = fmaxf(nm, -v);
    }
    r1[threadIdx.x] = mx; r2[threadIdx.x] = nm; __syncthreads();
    for (int st = 128; st > 0; st >>= 1) {
        if (threadIdx.x < st) {
            r1[threadIdx.x] = fmaxf(r1[threadIdx.x], r1[threadIdx.x + st]);
            r2[threadIdx.x] = fmaxf(r2[threadIdx.x], r2[threadIdx.x + st]);
        }
        __syncthreads();
    }
    if (threadIdx.x == 0) { atomicMax(mxslot, r1[0]); atomicMax(negmnslot, r2[0]); }
}

__global__ void qat_fq_asym_apply(float* __restrict__ base, int rows, int cols, int stride,
                                  const float* mxslot, const float* negmnslot) {
    float mx = *mxslot, mn = -*negmnslot;
    float s  = fmaxf((mx - mn) * (1.0f / 255.0f), 1e-8f);
    float zp = rintf(-mn / s);
    size_t n = (size_t)rows * cols;
    for (size_t i = (size_t)blockIdx.x * 256 + threadIdx.x; i < n; i += (size_t)gridDim.x * 256) {
        int r = (int)(i / cols), c = (int)(i % cols);
        float* p = base + (size_t)r * stride + c;
        float q = fminf(fmaxf(rintf(*p / s) + zp, 0.0f), 255.0f);
        *p = (q - zp) * s;
    }
}

// mid[M,16] = X[M,K] @ Aw[K,16]
__global__ void qat_lora_mid(const float* __restrict__ X, const float* __restrict__ Aw,
                             float* __restrict__ mid, int K) {
    __shared__ float red[256];
    int row = blockIdx.x, t = threadIdx.x;
    int r = t & 15, chunk = t >> 4;                  // t = chunk*16 + r
    float p = 0.0f;
    for (int k = chunk; k < K; k += 16)
        p += X[(size_t)row * K + k] * Aw[(size_t)k * 16 + r];
    red[t] = p; __syncthreads();
    if (t < 16) {
        float sum = 0.0f;
        for (int c = 0; c < 16; ++c) sum += red[c * 16 + t];
        mid[(size_t)row * 16 + t] = sum;
    }
}

// Out[M,N] += 2.0 * mid[M,16] @ Bw[16,N]
__global__ void qat_lora_add(const float* __restrict__ mid, const float* __restrict__ Bw,
                             float* __restrict__ Out, int Mr, int N) {
    size_t i = (size_t)blockIdx.x * 256 + threadIdx.x;
    if (i >= (size_t)Mr * N) return;
    int row = (int)(i / N), col = (int)(i % N);
    float sum = 0.0f;
    #pragma unroll
    for (int r = 0; r < 16; ++r)
        sum += mid[(size_t)row * 16 + r] * Bw[(size_t)r * N + col];
    Out[i] += sum * 2.0f;                            // SCALING = 32/16
}

__global__ void qat_resadd(float* __restrict__ h, const float* __restrict__ t1, size_t n) {
    size_t i = (size_t)blockIdx.x * 256 + threadIdx.x;
    if (i < n) h[i] += t1[i];
}

__global__ void qat_gelu(float* __restrict__ y, size_t n) {
    size_t i = (size_t)blockIdx.x * 256 + threadIdx.x;
    if (i < n) {
        float v = y[i];
        y[i] = 0.5f * v * (1.0f + erff(v * 0.70710678118654752f));
    }
}

// Causal attention, two-pass softmax per (b,h,q). qkv layout: [B*T, 2304],
// q at col h*64, k at 768+h*64, v at 1536+h*64. out merged [B*T, 768].
__global__ __launch_bounds__(64) void qat_attn(const float* __restrict__ qkv,
                                               float* __restrict__ o) {
    __shared__ float qs[64];
    __shared__ float sc[1024];
    __shared__ float red[64];
    int b = blockIdx.z, hh = blockIdx.y, qi = blockIdx.x;
    int t = threadIdx.x;
    const size_t rs = 2304;
    const float* qrow = qkv + (size_t)(b * 1024 + qi) * rs + hh * 64;
    qs[t] = qrow[t];
    __syncthreads();
    for (int k = t; k <= qi; k += 64) {
        const float* krow = qkv + (size_t)(b * 1024 + k) * rs + 768 + hh * 64;
        float s = 0.0f;
        for (int d = 0; d < 64; ++d) s += qs[d] * krow[d];
        sc[k] = s * 0.125f;                          // 1/sqrt(64)
    }
    __syncthreads();
    float m = -3.0e38f;
    for (int k = t; k <= qi; k += 64) m = fmaxf(m, sc[k]);
    red[t] = m; __syncthreads();
    for (int st = 32; st > 0; st >>= 1) { if (t < st) red[t] = fmaxf(red[t], red[t + st]); __syncthreads(); }
    m = red[0]; __syncthreads();
    float psum = 0.0f;
    for (int k = t; k <= qi; k += 64) { float e = __expf(sc[k] - m); sc[k] = e; psum += e; }
    red[t] = psum; __syncthreads();
    for (int st = 32; st > 0; st >>= 1) { if (t < st) red[t] += red[t + st]; __syncthreads(); }
    float denom = red[0]; __syncthreads();
    float acc = 0.0f;
    for (int k = 0; k <= qi; ++k)
        acc += sc[k] * qkv[(size_t)(b * 1024 + k) * rs + 1536 + hh * 64 + t];
    o[(size_t)(b * 1024 + qi) * 768 + hh * 64 + t] = acc / denom;
}

// ---------------------------------------------------------------------------
// Orchestration
// ---------------------------------------------------------------------------
extern "C" void kernel_launch(void* const* d_in, const int* in_sizes, int n_in,
                              void* d_out, int out_size, void* d_ws, size_t ws_size,
                              hipStream_t stream) {
    (void)in_sizes; (void)n_in; (void)out_size; (void)ws_size;
    const int*   ids  = (const int*)  d_in[0];
    const float* wte  = (const float*)d_in[1];
    const float* wpe  = (const float*)d_in[2];
    const float* ln1g = (const float*)d_in[3];
    const float* ln1b = (const float*)d_in[4];
    const float* qkvw = (const float*)d_in[5];
    const float* qkvb = (const float*)d_in[6];
    const float* qkvA = (const float*)d_in[7];
    const float* qkvB = (const float*)d_in[8];
    const float* aow  = (const float*)d_in[9];
    const float* aob  = (const float*)d_in[10];
    const float* aoA  = (const float*)d_in[11];
    const float* aoB  = (const float*)d_in[12];
    const float* ln2g = (const float*)d_in[13];
    const float* ln2b = (const float*)d_in[14];
    const float* fcw  = (const float*)d_in[15];
    const float* fcb  = (const float*)d_in[16];
    const float* fcA  = (const float*)d_in[17];
    const float* fcB  = (const float*)d_in[18];
    const float* mow  = (const float*)d_in[19];
    const float* mob  = (const float*)d_in[20];
    const float* moA  = (const float*)d_in[21];
    const float* moB  = (const float*)d_in[22];
    const float* lnfg = (const float*)d_in[23];
    const float* lnfb = (const float*)d_in[24];
    float* out = (float*)d_out;

    const int M = 2048, Dm = 768, R = 16, V = 50257;
    float* ws     = (float*)d_ws;
    float* scales = ws;                               // 256 slots
    float* h      = ws + 256;                         // [M,768]
    float* x      = h    + (size_t)M * Dm;            // [M,768]
    float* qkv    = x    + (size_t)M * Dm;            // [M,2304]
    float* obuf   = qkv  + (size_t)M * 3 * Dm;        // [M,768]
    float* t1     = obuf + (size_t)M * Dm;            // [M,3072]
    float* mid    = t1   + (size_t)M * 4 * Dm;        // [M,16]

    qat_init_scales<<<1, 256, 0, stream>>>(scales);
    qat_embed<<<M, 256, 0, stream>>>(ids, wte, wpe, h);

    const dim3 gQKV(18, 16), gD(6, 16), gFC(24, 16), gLM((V + 127) / 128, 16);

    for (int l = 0; l < 12; ++l) {
        float* sl = scales + l * 8;
        // ---- attn sublayer ----
        qat_ln<<<M, 256, 0, stream>>>(h, ln1g + l * Dm, ln1b + l * Dm, x, Dm);
        qat_maxabs<<<256, 256, 0, stream>>>(qkvw + (size_t)l * Dm * 3 * Dm, (size_t)Dm * 3 * Dm, sl + 0);
        qat_gemm_wmma<<<gQKV, 256, 0, stream>>>(x, qkvw + (size_t)l * Dm * 3 * Dm, sl + 0,
                                                qkvb + (size_t)l * 3 * Dm, qkv,
                                                M, 3 * Dm, Dm, 3 * Dm, 0, 1);
        qat_lora_mid<<<M, 256, 0, stream>>>(x, qkvA + (size_t)l * Dm * R, mid, Dm);
        qat_lora_add<<<(M * 3 * Dm + 255) / 256, 256, 0, stream>>>(mid, qkvB + (size_t)l * R * 3 * Dm,
                                                                   qkv, M, 3 * Dm);
        qat_minmax_strided<<<256, 256, 0, stream>>>(qkv + Dm,     M, Dm, 3 * Dm, sl + 4, sl + 5);
        qat_fq_asym_apply <<<256, 256, 0, stream>>>(qkv + Dm,     M, Dm, 3 * Dm, sl + 4, sl + 5);
        qat_minmax_strided<<<256, 256, 0, stream>>>(qkv + 2 * Dm, M, Dm, 3 * Dm, sl + 6, sl + 7);
        qat_fq_asym_apply <<<256, 256, 0, stream>>>(qkv + 2 * Dm, M, Dm, 3 * Dm, sl + 6, sl + 7);
        qat_attn<<<dim3(1024, 12, 2), 64, 0, stream>>>(qkv, obuf);
        qat_maxabs<<<256, 256, 0, stream>>>(aow + (size_t)l * Dm * Dm, (size_t)Dm * Dm, sl + 1);
        qat_gemm_wmma<<<gD, 256, 0, stream>>>(obuf, aow + (size_t)l * Dm * Dm, sl + 1,
                                              aob + (size_t)l * Dm, t1, M, Dm, Dm, Dm, 0, 1);
        qat_lora_mid<<<M, 256, 0, stream>>>(obuf, aoA + (size_t)l * Dm * R, mid, Dm);
        qat_lora_add<<<(M * Dm + 255) / 256, 256, 0, stream>>>(mid, aoB + (size_t)l * R * Dm, t1, M, Dm);
        qat_resadd<<<(M * Dm + 255) / 256, 256, 0, stream>>>(h, t1, (size_t)M * Dm);
        // ---- mlp sublayer ----
        qat_ln<<<M, 256, 0, stream>>>(h, ln2g + l * Dm, ln2b + l * Dm, x, Dm);
        qat_maxabs<<<256, 256, 0, stream>>>(fcw + (size_t)l * Dm * 4 * Dm, (size_t)Dm * 4 * Dm, sl + 2);
        qat_gemm_wmma<<<gFC, 256, 0, stream>>>(x, fcw + (size_t)l * Dm * 4 * Dm, sl + 2,
                                               fcb + (size_t)l * 4 * Dm, t1,
                                               M, 4 * Dm, Dm, 4 * Dm, 0, 1);
        qat_lora_mid<<<M, 256, 0, stream>>>(x, fcA + (size_t)l * Dm * R, mid, Dm);
        qat_lora_add<<<(M * 4 * Dm + 255) / 256, 256, 0, stream>>>(mid, fcB + (size_t)l * R * 4 * Dm,
                                                                   t1, M, 4 * Dm);
        qat_gelu<<<(M * 4 * Dm + 255) / 256, 256, 0, stream>>>(t1, (size_t)M * 4 * Dm);
        qat_maxabs<<<256, 256, 0, stream>>>(mow + (size_t)l * 4 * Dm * Dm, (size_t)4 * Dm * Dm, sl + 3);
        qat_gemm_wmma<<<gD, 256, 0, stream>>>(t1, mow + (size_t)l * 4 * Dm * Dm, sl + 3,
                                              mob + (size_t)l * Dm, obuf, M, Dm, 4 * Dm, Dm, 0, 1);
        qat_lora_mid<<<M, 256, 0, stream>>>(t1, moA + (size_t)l * 4 * Dm * R, mid, 4 * Dm);
        qat_lora_add<<<(M * Dm + 255) / 256, 256, 0, stream>>>(mid, moB + (size_t)l * R * Dm, obuf, M, Dm);
        qat_resadd<<<(M * Dm + 255) / 256, 256, 0, stream>>>(h, obuf, (size_t)M * Dm);
    }

    qat_ln<<<M, 256, 0, stream>>>(h, lnfg, lnfb, x, Dm);
    // logits = x @ wte.T  (no quant, no bias, W transposed)
    qat_gemm_wmma<<<gLM, 256, 0, stream>>>(x, wte, nullptr, nullptr, out,
                                           M, V, Dm, Dm, 1, 0);
}
